// MPNN_Surrogate_824633721181
// MI455X (gfx1250) — matrix operand order repaired and checked by reading, not compiled, not run on hardware
//
#include <hip/hip_runtime.h>
#include <hip/hip_bf16.h>

// ---------------------------------------------------------------------------
// MPNN surrogate for MI455X (gfx1250, wave32).
// h/agg live in workspace; edge+node MLPs are fused WMMA pipelines using
// v_wmma_f32_16x16x32_bf16 (16x16 C tiles, fp32 accumulate), per-wave LDS
// transpose between the two GEMMs, and f32 global atomics for scatter-add.
// An opaque zero offset (asm volatile) inside the tile loop stops LICM from
// hoisting the B-operand loads (previously a ~768-VGPR scratch spill) while
// keeping the pointers in the global address space (global_load_b128, not
// flat_load).  B loads for a whole K-chunk are batched ahead of the 8 WMMAs
// so the compiler can issue staggered partial s_wait_loadcnt waits.
// ---------------------------------------------------------------------------

typedef __bf16 bf16_t;
typedef __attribute__((ext_vector_type(16))) __bf16 v16bf;
typedef __attribute__((ext_vector_type(8)))  __bf16 v8bf;
typedef __attribute__((ext_vector_type(8)))  float  v8f;

#define HID 128

static __device__ __forceinline__ v16bf cat8(v8bf lo, v8bf hi) {
  return __builtin_shufflevector(lo, hi, 0,1,2,3,4,5,6,7,8,9,10,11,12,13,14,15);
}

// A-operand chunk (16 bf16 per lane) from a row-major bf16 row pointer.
// kloA = (lane<16 ? 0 : 8).  Loads row[klo..klo+7] and row[klo+16..klo+23].
static __device__ __forceinline__ v16bf load_a_bf16(const bf16_t* __restrict__ row, int kloA) {
  v8bf lo = *(const v8bf*)(row + kloA);
  v8bf hi = *(const v8bf*)(row + kloA + 16);
  return cat8(lo, hi);
}

// 8 contiguous f32 -> 8 bf16 (for A chunks sourced from the fp32 agg buffer)
static __device__ __forceinline__ v8bf cvt8f(const float* __restrict__ p) {
  float4 a = *(const float4*)p;
  float4 b = *(const float4*)(p + 4);
  v8bf r;
  r[0] = (bf16_t)a.x; r[1] = (bf16_t)a.y; r[2] = (bf16_t)a.z; r[3] = (bf16_t)a.w;
  r[4] = (bf16_t)b.x; r[5] = (bf16_t)b.y; r[6] = (bf16_t)b.z; r[7] = (bf16_t)b.w;
  return r;
}

// B-operand chunk from transposed weights wT[N][K]: lane's column is fixed,
// K values kb+8*(lane>=16)+[0..7] and kb+16+8*(lane>=16)+[0..7].
static __device__ __forceinline__ v16bf load_b(const bf16_t* __restrict__ wT_col_k) {
  v8bf lo = *(const v8bf*)(wT_col_k);
  v8bf hi = *(const v8bf*)(wT_col_k + 16);
  return cat8(lo, hi);
}

static __device__ __forceinline__ v8f wmma_bf16(v16bf a, v16bf b, v8f c) {
  return __builtin_amdgcn_wmma_f32_16x16x32_bf16(false, a, false, b, (short)0, c, false, false);
}

static __device__ __forceinline__ v8f splat8(float v) {
  v8f r;
#pragma unroll
  for (int i = 0; i < 8; ++i) r[i] = v;
  return r;
}

// Opaque zero: produced by volatile asm inside the tile loop, so pointers
// offset by it look loop-variant (no LICM of the weight loads), while the
// base pointer keeps its inferred global address space.
static __device__ __forceinline__ int opq() {
  int z = 0;
  asm volatile("" : "+s"(z));
  return z;
}

// ---------------------------------------------------------------------------
// Encoder: h = x @ enc_W + enc_b  (also writes bf16 mirror of h)
// ---------------------------------------------------------------------------
__global__ void __launch_bounds__(256)
encoder_kernel(const float* __restrict__ x, const float* __restrict__ W,
               const float* __restrict__ b, float* __restrict__ hf,
               bf16_t* __restrict__ hb, int nNodes) {
  int i = blockIdx.x * blockDim.x + threadIdx.x;
  if (i >= nNodes * HID) return;
  int n = i >> 7, j = i & (HID - 1);
  float acc = b[j];
  const float* xr = x + (size_t)n * 16;
#pragma unroll
  for (int k = 0; k < 16; ++k) acc += xr[k] * W[k * HID + j];
  hf[i] = acc;
  hb[i] = (bf16_t)acc;
}

// ---------------------------------------------------------------------------
// Weight prep: transpose all layer weights to bf16 [N][K] layout.
// ---------------------------------------------------------------------------
__global__ void __launch_bounds__(256)
prep_weights(const float* __restrict__ eW1, const float* __restrict__ eW2,
             const float* __restrict__ nW1, const float* __restrict__ nW2,
             bf16_t* __restrict__ eW1T, bf16_t* __restrict__ eW2T,
             bf16_t* __restrict__ nW1T, bf16_t* __restrict__ nW2T) {
  int i = blockIdx.x * blockDim.x + threadIdx.x;
  if (i < 3 * 256 * HID) {               // W1: [3][256][128] -> [3][128][256]
    int l = i / (256 * HID), rem = i % (256 * HID);
    int k = rem / HID, n = rem % HID;
    eW1T[l * 256 * HID + n * 256 + k] = (bf16_t)eW1[i];
    nW1T[l * 256 * HID + n * 256 + k] = (bf16_t)nW1[i];
  }
  if (i < 3 * HID * HID) {               // W2: [3][128][128] -> [3][128][128]^T
    int l = i / (HID * HID), rem = i % (HID * HID);
    int k = rem / HID, n = rem % HID;
    eW2T[l * HID * HID + n * HID + k] = (bf16_t)eW2[i];
    nW2T[l * HID * HID + n * HID + k] = (bf16_t)nW2[i];
  }
}

// ---------------------------------------------------------------------------
// Edge kernel: one wave = 16 edges.  msg = relu([h[row]||h[col]]W1+b1)W2+b2,
// scatter-added into agg with global f32 atomics.  96 WMMA per tile.
// ---------------------------------------------------------------------------
__global__ void __launch_bounds__(256)
edge_kernel(const int* __restrict__ ei, const bf16_t* __restrict__ hb,
            const bf16_t* __restrict__ w1T_in, const float* __restrict__ b1,
            const bf16_t* __restrict__ w2T_in, const float* __restrict__ b2,
            float* __restrict__ agg, int nEdges) {
  __shared__ bf16_t scratch[8][16 * HID];   // 4KB per wave
  const int lane   = threadIdx.x & 31;
  const int wid    = threadIdx.x >> 5;
  const int laneN  = lane & 15;
  const int hiHalf = lane >> 4;             // 0 or 1
  const int kloA   = hiHalf * 8;            // A-operand K base within chunk
  const int kB     = hiHalf * 8;            // B-operand K base within chunk
  const int nTiles = nEdges >> 4;

  for (int tile = blockIdx.x * 8 + wid; tile < nTiles; tile += gridDim.x * 8) {
    // per-lane weight base: column fixed (laneN), opaque zero defeats LICM
    const bf16_t* w1T = w1T_in + (size_t)laneN * 256 + kB + opq();
    const bf16_t* w2T = w2T_in + (size_t)laneN * HID + kB + opq();

    const int e0 = tile << 4;
    // A-row sources for this lane's matrix row (edge = e0 + laneN)
    const int rIdx = ei[e0 + laneN];             // src node (K 0..127)
    const int cIdx = ei[nEdges + e0 + laneN];    // dst node (K 128..255)
    const bf16_t* srcRow = hb + (size_t)rIdx * HID;
    const bf16_t* dstRow = hb + (size_t)cIdx * HID;

    // ---- GEMM1: [16x256] x [256x128], K in 8 chunks of 32 -----------------
    v8f acc[8];
#pragma unroll
    for (int t = 0; t < 8; ++t) acc[t] = splat8(b1[t * 16 + laneN]);
#pragma unroll
    for (int c = 0; c < 8; ++c) {
      const int kb = c * 32;
      const bf16_t* ar = (kb < HID) ? (srcRow + kb) : (dstRow + (kb - HID));
      v16bf a = load_a_bf16(ar, kloA);
      v16bf bm[8];
#pragma unroll
      for (int t = 0; t < 8; ++t) bm[t] = load_b(w1T + t * (16 * 256) + kb);
#pragma unroll
      for (int t = 0; t < 8; ++t) acc[t] = wmma_bf16(a, bm[t], acc[t]);
    }

    // ---- ReLU + transpose C-layout -> A-layout via per-wave LDS -----------
    bf16_t* sc = &scratch[wid][0];
#pragma unroll
    for (int t = 0; t < 8; ++t) {
#pragma unroll
      for (int r = 0; r < 8; ++r) {
        float v = acc[t][r];
        v = v > 0.f ? v : 0.f;
        const int m = hiHalf * 8 + r;
        sc[m * HID + t * 16 + laneN] = (bf16_t)v;
      }
    }
    __asm volatile("s_wait_dscnt 0x0" ::: "memory");

    // ---- GEMM2: [16x128] x [128x128], K in 4 chunks of 32 -----------------
    v8f acc2[8];
#pragma unroll
    for (int t = 0; t < 8; ++t) acc2[t] = splat8(b2[t * 16 + laneN]);
#pragma unroll
    for (int c = 0; c < 4; ++c) {
      const int kb = c * 32;
      v16bf a = load_a_bf16(sc + laneN * HID + kb, kloA);
      v16bf bm[8];
#pragma unroll
      for (int t = 0; t < 8; ++t) bm[t] = load_b(w2T + t * (16 * HID) + kb);
#pragma unroll
      for (int t = 0; t < 8; ++t) acc2[t] = wmma_bf16(a, bm[t], acc2[t]);
    }

    // ---- scatter-add messages into agg (global f32 atomics) ---------------
    int colIdx[8];
#pragma unroll
    for (int r = 0; r < 8; ++r) colIdx[r] = ei[nEdges + e0 + hiHalf * 8 + r];
#pragma unroll
    for (int t = 0; t < 8; ++t) {
#pragma unroll
      for (int r = 0; r < 8; ++r) {
        unsafeAtomicAdd(agg + (size_t)colIdx[r] * HID + t * 16 + laneN, acc2[t][r]);
      }
    }
  }
}

// ---------------------------------------------------------------------------
// Node kernel: one wave = 16 nodes.  upd = relu([h||agg]W1+b1)W2+b2;
// h += upd (fp32 master + bf16 mirror refresh).
// ---------------------------------------------------------------------------
__global__ void __launch_bounds__(256)
node_kernel(bf16_t* __restrict__ hb, const float* __restrict__ aggf,
            const bf16_t* __restrict__ w1T_in, const float* __restrict__ b1,
            const bf16_t* __restrict__ w2T_in, const float* __restrict__ b2,
            float* __restrict__ hf, int nNodes) {
  __shared__ bf16_t scratch[8][16 * HID];
  const int lane   = threadIdx.x & 31;
  const int wid    = threadIdx.x >> 5;
  const int laneN  = lane & 15;
  const int hiHalf = lane >> 4;
  const int kloA   = hiHalf * 8;
  const int kB     = hiHalf * 8;
  const int nTiles = nNodes >> 4;

  for (int tile = blockIdx.x * 8 + wid; tile < nTiles; tile += gridDim.x * 8) {
    const bf16_t* w1T = w1T_in + (size_t)laneN * 256 + kB + opq();
    const bf16_t* w2T = w2T_in + (size_t)laneN * HID + kB + opq();

    const int base = tile << 4;
    const int myRow = base + laneN;
    const bf16_t* hRow   = hb   + (size_t)myRow * HID;
    const float*  aggRow = aggf + (size_t)myRow * HID;

    // ---- GEMM1: A = [h_bf16 || bf16(agg)], K = 256 ------------------------
    v8f acc[8];
#pragma unroll
    for (int t = 0; t < 8; ++t) acc[t] = splat8(b1[t * 16 + laneN]);
#pragma unroll
    for (int c = 0; c < 8; ++c) {
      const int kb = c * 32;
      v16bf a;
      if (kb < HID) {
        a = load_a_bf16(hRow + kb, kloA);
      } else {
        const float* p = aggRow + (kb - HID) + kloA;
        a = cat8(cvt8f(p), cvt8f(p + 16));
      }
      v16bf bm[8];
#pragma unroll
      for (int t = 0; t < 8; ++t) bm[t] = load_b(w1T + t * (16 * 256) + kb);
#pragma unroll
      for (int t = 0; t < 8; ++t) acc[t] = wmma_bf16(a, bm[t], acc[t]);
    }

    // ---- ReLU + LDS transpose --------------------------------------------
    bf16_t* sc = &scratch[wid][0];
#pragma unroll
    for (int t = 0; t < 8; ++t) {
#pragma unroll
      for (int r = 0; r < 8; ++r) {
        float v = acc[t][r];
        v = v > 0.f ? v : 0.f;
        const int m = hiHalf * 8 + r;
        sc[m * HID + t * 16 + laneN] = (bf16_t)v;
      }
    }
    __asm volatile("s_wait_dscnt 0x0" ::: "memory");

    // ---- GEMM2 ------------------------------------------------------------
    v8f acc2[8];
#pragma unroll
    for (int t = 0; t < 8; ++t) acc2[t] = splat8(b2[t * 16 + laneN]);
#pragma unroll
    for (int c = 0; c < 4; ++c) {
      const int kb = c * 32;
      v16bf a = load_a_bf16(sc + laneN * HID + kb, kloA);
      v16bf bm[8];
#pragma unroll
      for (int t = 0; t < 8; ++t) bm[t] = load_b(w2T + t * (16 * HID) + kb);
#pragma unroll
      for (int t = 0; t < 8; ++t) acc2[t] = wmma_bf16(a, bm[t], acc2[t]);
    }

    // ---- residual update: h += upd, refresh bf16 mirror -------------------
#pragma unroll
    for (int t = 0; t < 8; ++t) {
#pragma unroll
      for (int r = 0; r < 8; ++r) {
        const int m = hiHalf * 8 + r;
        const size_t idx = (size_t)(base + m) * HID + t * 16 + laneN;
        float nh = hf[idx] + acc2[t][r];
        hf[idx] = nh;
        hb[idx] = (bf16_t)nh;
      }
    }
  }
}

// ---------------------------------------------------------------------------
// Decoder: out = h @ dec_W + dec_b   [N,128] x [128,3]
// ---------------------------------------------------------------------------
__global__ void __launch_bounds__(256)
decoder_kernel(const float* __restrict__ hf, const float* __restrict__ W,
               const float* __restrict__ b, float* __restrict__ out, int nNodes) {
  int i = blockIdx.x * blockDim.x + threadIdx.x;
  if (i >= nNodes * 3) return;
  int n = i / 3, j = i % 3;
  float acc = b[j];
  const float* hr = hf + (size_t)n * HID;
#pragma unroll 8
  for (int k = 0; k < HID; ++k) acc += hr[k] * W[k * 3 + j];
  out[i] = acc;
}

// ---------------------------------------------------------------------------
extern "C" void kernel_launch(void* const* d_in, const int* in_sizes, int n_in,
                              void* d_out, int out_size, void* d_ws, size_t ws_size,
                              hipStream_t stream) {
  const float* x    = (const float*)d_in[0];
  const int*   ei   = (const int*)d_in[1];
  const float* encW = (const float*)d_in[2];
  const float* encb = (const float*)d_in[3];
  const float* decW = (const float*)d_in[4];
  const float* decb = (const float*)d_in[5];
  const float* eW1  = (const float*)d_in[6];
  const float* eb1  = (const float*)d_in[7];
  const float* eW2  = (const float*)d_in[8];
  const float* eb2  = (const float*)d_in[9];
  const float* nW1  = (const float*)d_in[10];
  const float* nb1  = (const float*)d_in[11];
  const float* nW2  = (const float*)d_in[12];
  const float* nb2  = (const float*)d_in[13];

  const int nNodes = in_sizes[0] / 16;   // 30000
  const int nEdges = in_sizes[1] / 2;    // 480000

  // workspace layout (all offsets 16B aligned)
  char*   ws   = (char*)d_ws;
  float*  hf   = (float*)ws;                                          // N*128 f32
  bf16_t* hb   = (bf16_t*)(ws + (size_t)nNodes * HID * 4);            // N*128 bf16
  float*  agg  = (float*)(ws + (size_t)nNodes * HID * 6);             // N*128 f32
  bf16_t* wbuf = (bf16_t*)(ws + (size_t)nNodes * HID * 10);
  bf16_t* eW1T = wbuf;                       // 3*128*256
  bf16_t* eW2T = eW1T + 3 * 256 * HID;       // 3*128*128
  bf16_t* nW1T = eW2T + 3 * HID * HID;
  bf16_t* nW2T = nW1T + 3 * 256 * HID;

  encoder_kernel<<<(nNodes * HID + 255) / 256, 256, 0, stream>>>(x, encW, encb, hf, hb, nNodes);
  prep_weights<<<(3 * 256 * HID + 255) / 256, 256, 0, stream>>>(eW1, eW2, nW1, nW2,
                                                                eW1T, eW2T, nW1T, nW2T);

  const int edgeBlocks = (nEdges / 16 + 7) / 8;   // one wave per 16-edge tile
  const int nodeBlocks = (nNodes / 16 + 7) / 8;
  for (int l = 0; l < 3; ++l) {
    hipMemsetAsync(agg, 0, (size_t)nNodes * HID * 4, stream);
    edge_kernel<<<edgeBlocks, 256, 0, stream>>>(ei, hb,
        eW1T + (size_t)l * 256 * HID, eb1 + l * HID,
        eW2T + (size_t)l * HID * HID, eb2 + l * HID, agg, nEdges);
    node_kernel<<<nodeBlocks, 256, 0, stream>>>(hb, agg,
        nW1T + (size_t)l * 256 * HID, nb1 + l * HID,
        nW2T + (size_t)l * HID * HID, nb2 + l * HID, hf, nNodes);
  }
  decoder_kernel<<<(nNodes * 3 + 255) / 256, 256, 0, stream>>>(hf, decW, decb,
                                                               (float*)d_out, nNodes);
}
